// GNNCritic_30382598652559
// MI455X (gfx1250) — matrix operand order repaired and checked by reading, not compiled, not run on hardware
//
#include <hip/hip_runtime.h>
#include <math.h>

typedef __attribute__((ext_vector_type(16))) _Float16 v16h;
typedef __attribute__((ext_vector_type(8)))  float    v8f;

#define HS    64
#define NPRED 5
#define NPREY 50
#define NOBST 10
#define NTOK  65
#define KP    96   // padded K for xp (66 -> 96)

// ---- LDS layout (units: halves) ----
#define OFF_SX   0                    // sx[80][96]  embeddings + pos, zero padded (rows 65..79 stay zero)
#define OFF_WQT  (OFF_SX  + 80*96)    // WqT[64][96]
#define OFF_WKT  (OFF_WQT + 64*96)    // WkT[64][96]
#define OFF_WVT  (OFF_WKT + 64*96)    // WvT[64][64]
#define OFF_WOT  (OFF_WVT + 64*64)    // WoT[64][64]
#define OFF_W1T  (OFF_WOT + 64*64)    // W1T[64][64]
#define OFF_W2T  (OFF_W1T + 64*64)    // W2T[64][64]
#define OFF_KB   (OFF_W2T + 64*64)    // kbuf[80][64] row-major
#define OFF_QB   (OFF_KB  + 80*64)    // qbuf[16][64]
#define OFF_VT   (OFF_QB  + 16*64)    // vT[64][96]  (dh-col major, key rows padded to 96)
#define OFF_AT   (OFF_VT  + 64*96)    // attn[4][16][96] f16, zero padded
#define OFF_CX   (OFF_AT  + 4*16*96)  // ctxo[16][64]
#define OFF_M0   (OFF_CX  + 16*64)    // mlp ping
#define OFF_M1   (OFF_M0  + 16*64)    // mlp pong
#define HALVES_TOTAL (OFF_M1 + 16*64)
#define LGTS_FLOATS  (4*16*80)        // logits f32 [4][16][80]
#define SMEM_BYTES   (HALVES_TOTAL*2 + LGTS_FLOATS*4)

#define ZROW_OFF (OFF_SX + 72*KP)     // a row of >=32 guaranteed-zero halves

__device__ __forceinline__ v8f wmma_f16(v16h a, v16h b, v8f c) {
  return __builtin_amdgcn_wmma_f32_16x16x32_f16(
      /*neg_a=*/false, a, /*neg_b=*/false, b,
      /*c_mod=*/(short)0, c, /*reuse_a=*/false, /*reuse_b=*/false);
}

// A-tile (16x32 f16) from row-major LDS, row stride ld (halves).
// lane l: m = l&15, hi = l>>4; elem e -> K = k0+hi*16+e
__device__ __forceinline__ v16h load_a16(const _Float16* s, int ld, int r0, int k0) {
  int l = threadIdx.x & 31, m = l & 15, hi = l >> 4;
  const _Float16* p = s + (r0 + m) * ld + k0 + hi * 16;
  v16h a;
#pragma unroll
  for (int e = 0; e < 16; ++e) a[e] = p[e];
  return a;
}

// B-tile (32x16 f16) from TRANSPOSED storage srcT[n][k], row stride ld.
// lane l: n = l&15, hi = l>>4; elem e -> K = k0+hi*16+e
__device__ __forceinline__ v16h load_bT(const _Float16* sT, int ld, int n0, int k0) {
  int l = threadIdx.x & 31, n = l & 15, hi = l >> 4;
  const _Float16* p = sT + (n0 + n) * ld + k0 + hi * 16;
  v16h b;
#pragma unroll
  for (int e = 0; e < 16; ++e) b[e] = p[e];
  return b;
}

// q as A with dh=16 zero-padded to K=32: lanes 16..31 (K>=16) are zero
__device__ __forceinline__ v16h load_a_q(const _Float16* qb, int h) {
  int l = threadIdx.x & 31, m = l & 15, hi = l >> 4;
  const _Float16* p = qb + m * 64 + h * 16;
  v16h a;
#pragma unroll
  for (int e = 0; e < 16; ++e) a[e] = hi ? (_Float16)0.f : p[e];
  return a;
}

// k^T as B: B[kk][n] = kbuf[c0+n][h*16+kk]; lanes 16..31 (K>=16) read a zero LDS row
// (pointer select -> unconditional contiguous loads, no per-element cndmask)
__device__ __forceinline__ v16h load_b_kT(const _Float16* kb, const _Float16* zrow,
                                          int h, int c0) {
  int l = threadIdx.x & 31, n = l & 15, hi = l >> 4;
  const _Float16* p = hi ? zrow : (kb + (c0 + n) * 64 + h * 16);
  v16h b;
#pragma unroll
  for (int e = 0; e < 16; ++e) b[e] = p[e];
  return b;
}

__device__ __forceinline__ unsigned pack2h(float lo, float hi) {
  union { _Float16 h[2]; unsigned u; } pk;
  pk.h[0] = (_Float16)lo;
  pk.h[1] = (_Float16)hi;
  return pk.u;
}

__global__ __launch_bounds__(256)
void gnn_critic_fused(const float* __restrict__ pred, const float* __restrict__ prey,
                      const float* __restrict__ obst, const float* __restrict__ act,
                      const float* Wp, const float* bp, const float* Wy, const float* by_,
                      const float* Wob, const float* bob, const float* Wa, const float* ba,
                      const float* Wq, const float* bq, const float* Wk, const float* bk,
                      const float* Wv, const float* bv, const float* Wo, const float* bo,
                      const float* W1, const float* b1, const float* W2, const float* b2,
                      const float* W3, const float* b3, float* __restrict__ out)
{
  extern __shared__ char smem_raw[];
  _Float16* sh   = (_Float16*)smem_raw;
  _Float16* sx   = sh + OFF_SX;
  _Float16* wqT  = sh + OFF_WQT;
  _Float16* wkT  = sh + OFF_WKT;
  _Float16* wvT  = sh + OFF_WVT;
  _Float16* woT  = sh + OFF_WOT;
  _Float16* w1T  = sh + OFF_W1T;
  _Float16* w2T  = sh + OFF_W2T;
  _Float16* kbuf = sh + OFF_KB;
  _Float16* qbuf = sh + OFF_QB;
  _Float16* vT   = sh + OFF_VT;
  _Float16* attn = sh + OFF_AT;
  _Float16* ctxo = sh + OFF_CX;
  _Float16* mlp0 = sh + OFF_M0;
  _Float16* mlp1 = sh + OFF_M1;
  _Float16* zrow = sh + ZROW_OFF;
  float*    lgts = (float*)(sh + HALVES_TOTAL);

  const int t = threadIdx.x;
  const int wave = t >> 5;
  const int lane = t & 31;
  const int n16  = lane & 15;
  const int hi16 = lane >> 4;
  const int bidx = blockIdx.x;

  // ---------- Phase A: zero pads, weight convert, embeddings ----------
  for (int i = t; i < 80 * KP;     i += 256) sx[i]   = (_Float16)0.f;
  for (int i = t; i < 64 * KP;     i += 256) vT[i]   = (_Float16)0.f;
  for (int i = t; i < 4 * 16 * KP; i += 256) attn[i] = (_Float16)0.f;

  for (int i = t; i < 64 * KP; i += 256) {
    int n = i / KP, k = i % KP;
    float vq = (k < 66) ? Wq[k * HS + n] : 0.f;
    float vk = (k < 66) ? Wk[k * HS + n] : 0.f;
    wqT[i] = (_Float16)vq;
    wkT[i] = (_Float16)vk;
  }
  for (int i = t; i < 64 * 64; i += 256) {
    int n = i / 64, k = i % 64;
    wvT[i] = (_Float16)Wv[k * HS + n];
    woT[i] = (_Float16)Wo[k * HS + n];
    w1T[i] = (_Float16)W1[k * HS + n];
    w2T[i] = (_Float16)W2[k * HS + n];
  }

  for (int i = t; i < NTOK * HS; i += 256) {
    int r = i / HS, c = i % HS;
    float h;
    if (r < NPRED) {
      float s0 = pred[bidx * NPRED * 2 + r * 2];
      float s1 = pred[bidx * NPRED * 2 + r * 2 + 1];
      h = fmaxf(s0 * Wp[c] + s1 * Wp[HS + c] + bp[c], 0.f);
      float a = act[bidx * NPRED + r] * 3.14159265358979323846f;
      h += __sinf(a) * Wa[c] + __cosf(a) * Wa[HS + c] + ba[c];
    } else if (r < NPRED + NPREY) {
      int j = r - NPRED;
      float s0 = prey[bidx * NPREY * 2 + j * 2];
      float s1 = prey[bidx * NPREY * 2 + j * 2 + 1];
      h = fmaxf(s0 * Wy[c] + s1 * Wy[HS + c] + by_[c], 0.f);
    } else {
      int j = r - NPRED - NPREY;
      const float* o = obst + bidx * NOBST * 3 + j * 3;
      h = fmaxf(o[0] * Wob[c] + o[1] * Wob[HS + c] + o[2] * Wob[2 * HS + c] + bob[c], 0.f);
    }
    sx[r * KP + c] = (_Float16)h;
  }
  for (int r = t; r < NTOK; r += 256) {
    float px, py;
    if (r < NPRED)              { px = pred[bidx * 10 + r * 2];          py = pred[bidx * 10 + r * 2 + 1]; }
    else if (r < NPRED + NPREY) { int j = r - NPRED; px = prey[bidx * 100 + j * 2]; py = prey[bidx * 100 + j * 2 + 1]; }
    else                        { int j = r - NPRED - NPREY; px = obst[bidx * 30 + j * 3]; py = obst[bidx * 30 + j * 3 + 1]; }
    sx[r * KP + 64] = (_Float16)px;
    sx[r * KP + 65] = (_Float16)py;
  }
  __syncthreads();

  // ---------- Phase B1: k = xp @ Wk + bk  (20 tiles, K=96) ----------
  for (int job = wave; job < 20; job += 8) {
    int mt = job / 4, nt = job % 4;
    v8f c = {};
#pragma unroll
    for (int ks = 0; ks < 3; ++ks)
      c = wmma_f16(load_a16(sx, KP, mt * 16, ks * 32),
                   load_bT(wkT, KP, nt * 16, ks * 32), c);
    float bb = bk[nt * 16 + n16];
#pragma unroll
    for (int j = 0; j < 8; ++j)
      kbuf[(mt * 16 + j + hi16 * 8) * 64 + nt * 16 + n16] = (_Float16)(c[j] + bb);
  }

  // ---------- Phase B2: q = xp @ Wq + bq  (4 tiles, rows 0..15, waves 4..7) ----------
  if (wave >= 4) {
    int nt = wave - 4;
    v8f c = {};
#pragma unroll
    for (int ks = 0; ks < 3; ++ks)
      c = wmma_f16(load_a16(sx, KP, 0, ks * 32),
                   load_bT(wqT, KP, nt * 16, ks * 32), c);
    float bb = bq[nt * 16 + n16];
#pragma unroll
    for (int j = 0; j < 8; ++j)
      qbuf[(j + hi16 * 8) * 64 + nt * 16 + n16] = (_Float16)(c[j] + bb);
  }

  // ---------- Phase B3: v = x @ Wv + bv (20 tiles, K=64), stored transposed ----------
  for (int job = wave; job < 20; job += 8) {
    int mt = job / 4, nt = job % 4;
    v8f c = {};
#pragma unroll
    for (int ks = 0; ks < 2; ++ks)
      c = wmma_f16(load_a16(sx, KP, mt * 16, ks * 32),
                   load_bT(wvT, 64, nt * 16, ks * 32), c);
    float bb = bv[nt * 16 + n16];
    unsigned* dst = (unsigned*)&vT[(nt * 16 + n16) * KP + mt * 16 + hi16 * 8];
#pragma unroll
    for (int j = 0; j < 8; j += 2)
      dst[j >> 1] = pack2h(c[j] + bb, c[j + 1] + bb);
  }
  __syncthreads();

  // ---------- Phase C: logits = q @ k^T / sqrt(16) (only first M-tile) ----------
  for (int job = wave; job < 20; job += 8) {
    int h = job / 5, nt = job % 5;
    v8f c = {};
    c = wmma_f16(load_a_q(qbuf, h), load_b_kT(kbuf, zrow, h, nt * 16), c);
#pragma unroll
    for (int j = 0; j < 8; ++j)
      lgts[h * 16 * 80 + (j + hi16 * 8) * 80 + nt * 16 + n16] = c[j] * 0.25f;
  }
  __syncthreads();

  // ---------- Phase D: softmax over 65 keys, rows 0..4, 4 heads ----------
  for (int rr = wave; rr < 20; rr += 8) {
    int h = rr / 5, r = rr % 5;
    const float* row = lgts + h * 1280 + r * 80;
    float v0 = row[lane], v1 = row[lane + 32];
    float v2 = (lane == 0) ? row[64] : -1e30f;
    float mx = fmaxf(fmaxf(v0, v1), v2);
#pragma unroll
    for (int off = 16; off; off >>= 1) mx = fmaxf(mx, __shfl_xor(mx, off, 32));
    float e0 = __expf(v0 - mx), e1 = __expf(v1 - mx);
    float e2 = (lane == 0) ? __expf(v2 - mx) : 0.f;
    float s = e0 + e1 + e2;
#pragma unroll
    for (int off = 16; off; off >>= 1) s += __shfl_xor(s, off, 32);
    float inv = 1.f / s;
    _Float16* arow = attn + h * 16 * KP + r * KP;
    arow[lane]      = (_Float16)(e0 * inv);
    arow[lane + 32] = (_Float16)(e1 * inv);
    if (lane == 0) arow[64] = (_Float16)(e2 * inv);
  }
  __syncthreads();

  // ---------- Phase E: ctx = attn @ v (per head, K padded to 96 w/ exact zeros) ----------
  if (wave < 4) {
    int h = wave;
    v8f c = {};
#pragma unroll
    for (int ks = 0; ks < 3; ++ks)
      c = wmma_f16(load_a16(attn + h * 16 * KP, KP, 0, ks * 32),
                   load_bT(vT, KP, h * 16, ks * 32), c);
#pragma unroll
    for (int j = 0; j < 8; ++j)
      ctxo[(j + hi16 * 8) * 64 + h * 16 + n16] = (_Float16)c[j];
  }
  __syncthreads();

  // ---------- Phase F: out = x + ctx @ Wo + bo (first 16 rows only) ----------
  if (wave < 4) {
    int nt = wave;
    v8f c = {};
#pragma unroll
    for (int ks = 0; ks < 2; ++ks)
      c = wmma_f16(load_a16(ctxo, 64, 0, ks * 32),
                   load_bT(woT, 64, nt * 16, ks * 32), c);
#pragma unroll
    for (int j = 0; j < 8; ++j) {
      int m = j + hi16 * 8, col = nt * 16 + n16;
      float val = c[j] + bo[col] + (float)sx[m * KP + col];
      mlp0[m * 64 + col] = (_Float16)val;
    }
  }
  __syncthreads();

  // ---------- Phase G1: h1 = relu(out @ W1 + b1) ----------
  if (wave < 4) {
    int nt = wave;
    v8f c = {};
#pragma unroll
    for (int ks = 0; ks < 2; ++ks)
      c = wmma_f16(load_a16(mlp0, 64, 0, ks * 32),
                   load_bT(w1T, 64, nt * 16, ks * 32), c);
#pragma unroll
    for (int j = 0; j < 8; ++j) {
      int m = j + hi16 * 8, col = nt * 16 + n16;
      mlp1[m * 64 + col] = (_Float16)fmaxf(c[j] + b1[col], 0.f);
    }
  }
  __syncthreads();

  // ---------- Phase G2: h2 = relu(h1 @ W2 + b2) ----------
  if (wave < 4) {
    int nt = wave;
    v8f c = {};
#pragma unroll
    for (int ks = 0; ks < 2; ++ks)
      c = wmma_f16(load_a16(mlp1, 64, 0, ks * 32),
                   load_bT(w2T, 64, nt * 16, ks * 32), c);
#pragma unroll
    for (int j = 0; j < 8; ++j) {
      int m = j + hi16 * 8, col = nt * 16 + n16;
      mlp0[m * 64 + col] = (_Float16)fmaxf(c[j] + b2[col], 0.f);
    }
  }
  __syncthreads();

  // ---------- Phase H: y = h2 @ W3 + b3 (rows 0..4) ----------
  if (wave < NPRED) {
    int r = wave;
    float s = (float)mlp0[r * 64 + lane] * W3[lane]
            + (float)mlp0[r * 64 + lane + 32] * W3[lane + 32];
#pragma unroll
    for (int off = 16; off; off >>= 1) s += __shfl_xor(s, off, 32);
    if (lane == 0) out[bidx * NPRED + r] = s + b3[0];
  }
}

extern "C" void kernel_launch(void* const* d_in, const int* in_sizes, int n_in,
                              void* d_out, int out_size, void* d_ws, size_t ws_size,
                              hipStream_t stream) {
  const float* pred = (const float*)d_in[0];
  const float* prey = (const float*)d_in[1];
  const float* obst = (const float*)d_in[2];
  const float* act  = (const float*)d_in[3];
  const float* Wp = (const float*)d_in[4];  const float* bp  = (const float*)d_in[5];
  const float* Wy = (const float*)d_in[6];  const float* by_ = (const float*)d_in[7];
  const float* Wob = (const float*)d_in[8]; const float* bob = (const float*)d_in[9];
  const float* Wa = (const float*)d_in[10]; const float* ba  = (const float*)d_in[11];
  const float* Wq = (const float*)d_in[12]; const float* bq  = (const float*)d_in[13];
  const float* Wk = (const float*)d_in[14]; const float* bk  = (const float*)d_in[15];
  const float* Wv = (const float*)d_in[16]; const float* bv  = (const float*)d_in[17];
  const float* Wo = (const float*)d_in[18]; const float* bo  = (const float*)d_in[19];
  const float* W1 = (const float*)d_in[20]; const float* b1  = (const float*)d_in[21];
  const float* W2 = (const float*)d_in[22]; const float* b2  = (const float*)d_in[23];
  const float* W3 = (const float*)d_in[24]; const float* b3  = (const float*)d_in[25];
  float* out = (float*)d_out;

  int nB = in_sizes[0] / (NPRED * 2);  // 8192
  dim3 grid(nB), block(256);
  gnn_critic_fused<<<grid, block, SMEM_BYTES, stream>>>(
      pred, prey, obst, act, Wp, bp, Wy, by_, Wob, bob, Wa, ba,
      Wq, bq, Wk, bk, Wv, bv, Wo, bo, W1, b1, W2, b2, W3, b3, out);
}
